// GNNPredictor_65481071403354
// MI455X (gfx1250) — matrix operand order repaired and checked by reading, not compiled, not run on hardware
//
#include <hip/hip_runtime.h>

typedef __attribute__((ext_vector_type(16))) __bf16        v16bf;
typedef __attribute__((ext_vector_type(8)))  float         v8f;
typedef __attribute__((ext_vector_type(4)))  unsigned int  v4u;

union FragBF { v16bf v; v4u q[2]; };

#define CH 128          // all channel dims are 128
#define LDS_STRIDE 136  // 128 + 8 bf16 pad (272 B rows, 16B-aligned chunks)

__device__ inline void atomAddF(float* p, float v) { unsafeAtomicAdd(p, v); }

// ---------------------------------------------------------------------------
// T[r, :] = dinv[r] * (X[r, :] @ W)   computed with v_wmma_f32_16x16x32_bf16
// block = 256 threads (8 waves), each block does 128 rows x 128 cols.
// ---------------------------------------------------------------------------
__global__ __launch_bounds__(256) void gemm_rowscale_bf16(
    const float* __restrict__ X, const float* __restrict__ W,
    const float* __restrict__ dinv, float* __restrict__ T, int N)
{
    __shared__ __bf16 sX [CH][LDS_STRIDE];   // X tile, row-major (rows x K)
    __shared__ __bf16 sWt[CH][LDS_STRIDE];   // W transposed (cols x K)

    const int tid   = threadIdx.x;
    const int rbase = blockIdx.x * 128;

    // cooperative load: X tile (f32 -> bf16), zero-padded past N
    for (int i = tid; i < 128 * 32; i += 256) {
        int r = i >> 5, c4 = (i & 31) << 2;
        float4 v = make_float4(0.f, 0.f, 0.f, 0.f);
        if (rbase + r < N)
            v = *(const float4*)(X + (long long)(rbase + r) * CH + c4);
        sX[r][c4 + 0] = (__bf16)v.x; sX[r][c4 + 1] = (__bf16)v.y;
        sX[r][c4 + 2] = (__bf16)v.z; sX[r][c4 + 3] = (__bf16)v.w;
    }
    // cooperative load: W (K x C row-major in global) -> transposed bf16 in LDS
    for (int i = tid; i < 128 * 32; i += 256) {
        int k = i >> 5, c4 = (i & 31) << 2;
        float4 v = *(const float4*)(W + (long long)k * CH + c4);
        sWt[c4 + 0][k] = (__bf16)v.x; sWt[c4 + 1][k] = (__bf16)v.y;
        sWt[c4 + 2][k] = (__bf16)v.z; sWt[c4 + 3][k] = (__bf16)v.w;
    }
    __syncthreads();

    const int wave = tid >> 5, lane = tid & 31;
    const int m = lane & 15;       // row (A) / col (B) within 16
    const int h = lane >> 4;       // K half selector
    const int rowTile = wave * 16; // this wave's 16 output rows

    v8f acc[8] = {};               // 8 col-tiles of 16x16 f32

    for (int k0 = 0; k0 < CH; k0 += 32) {
        FragBF a;                  // A: row rowTile+m, K = k0+[8h,8h+8) U [16+8h,...)
        a.q[0] = *(const v4u*)&sX[rowTile + m][k0 + 8 * h];
        a.q[1] = *(const v4u*)&sX[rowTile + m][k0 + 16 + 8 * h];
        #pragma unroll
        for (int ct = 0; ct < 8; ++ct) {
            FragBF b;              // B: col ct*16+m, K = k0+[8h,8h+8) U [16+8h,...)
            b.q[0] = *(const v4u*)&sWt[ct * 16 + m][k0 + 8 * h];
            b.q[1] = *(const v4u*)&sWt[ct * 16 + m][k0 + 16 + 8 * h];
            acc[ct] = __builtin_amdgcn_wmma_f32_16x16x32_bf16(
                false, a.v, false, b.v, (short)0, acc[ct], false, false);
        }
    }

    // epilogue: D vgpr v <-> row 8h+v, lane%16 <-> col; scale rows by dinv
    #pragma unroll
    for (int v = 0; v < 8; ++v) {
        int r = rbase + rowTile + 8 * h + v;
        if (r < N) {
            float dv = dinv[r];
            #pragma unroll
            for (int ct = 0; ct < 8; ++ct)
                T[(long long)r * CH + ct * 16 + m] = dv * acc[ct][v];
        }
    }
}

// ---------------------------------------------------------------------------
// degree / normalization
// ---------------------------------------------------------------------------
__global__ void k_fill(float* p, float val, long long n) {
    long long i = (long long)blockIdx.x * blockDim.x + threadIdx.x;
    if (i < n) p[i] = val;
}
__global__ void k_deg(const long long* __restrict__ ei, long long E, float* deg) {
    long long e = (long long)blockIdx.x * blockDim.x + threadIdx.x;
    if (e < E) atomAddF(&deg[ei[E + e]], 1.0f);   // dst row of edge_index
}
__global__ void k_rsqrt(float* d, long long n) {
    long long i = (long long)blockIdx.x * blockDim.x + threadIdx.x;
    if (i < n) d[i] = rsqrtf(d[i]);
}

// ---------------------------------------------------------------------------
// edge scatter: A[dst] += T[src]; one wave per edge, float4 per lane (512B/row)
// ---------------------------------------------------------------------------
__global__ __launch_bounds__(256) void k_scatter(
    const long long* __restrict__ ei, long long E,
    const float* __restrict__ T, float* __restrict__ A)
{
    long long e = (long long)blockIdx.x * 8 + (threadIdx.x >> 5);
    if (e >= E) return;
    int lane = threadIdx.x & 31;
    long long s = ei[e];
    long long d = ei[E + e];
    float4 msg = *(const float4*)(T + s * CH + lane * 4);
    float* out = A + d * CH + lane * 4;
    atomAddF(out + 0, msg.x); atomAddF(out + 1, msg.y);
    atomAddF(out + 2, msg.z); atomAddF(out + 3, msg.w);
}

// H[r,c] = relu(dinv[r]*A[r,c] + bias[c])  (in place on A)
__global__ __launch_bounds__(256) void k_finalize(
    float* __restrict__ A, const float* __restrict__ dinv,
    const float* __restrict__ bias, long long nNodes)
{
    long long i = (long long)blockIdx.x * blockDim.x + threadIdx.x; // 4 elems/thread
    if (i >= nNodes * 32) return;
    long long r = i >> 5; int c4 = (int)(i & 31) << 2;
    float dv = dinv[r];
    float4 v = *(float4*)(A + r * CH + c4);
    const float4 b = *(const float4*)(bias + c4);
    v.x = fmaxf(fmaf(dv, v.x, b.x), 0.f);
    v.y = fmaxf(fmaf(dv, v.y, b.y), 0.f);
    v.z = fmaxf(fmaf(dv, v.z, b.z), 0.f);
    v.w = fmaxf(fmaf(dv, v.w, b.w), 0.f);
    *(float4*)(A + r * CH + c4) = v;
}

// column sums over nodes -> pooled[128]
__global__ __launch_bounds__(128) void k_pool(
    const float* __restrict__ H, long long nNodes, float* pooled)
{
    int c = threadIdx.x;
    long long r0 = (long long)blockIdx.x * 256;
    long long r1 = min(r0 + 256, nNodes);
    float s = 0.f;
    for (long long r = r0; r < r1; ++r) s += H[r * CH + c];
    atomAddF(&pooled[c], s);
}

// out[c] = bfc[c] + sum_k (pooled[k]/N) * Wfc[k,c]
__global__ __launch_bounds__(128) void k_fc(
    const float* __restrict__ pooled, const float* __restrict__ Wfc,
    const float* __restrict__ bfc, float* __restrict__ out, float invN)
{
    __shared__ float sp[CH];
    int c = threadIdx.x;
    sp[c] = pooled[c] * invN;
    __syncthreads();
    float s = bfc[c];
    #pragma unroll 8
    for (int k = 0; k < CH; ++k) s = fmaf(sp[k], Wfc[k * CH + c], s);
    out[c] = s;
}

// ---------------------------------------------------------------------------
extern "C" void kernel_launch(void* const* d_in, const int* in_sizes, int n_in,
                              void* d_out, int out_size, void* d_ws, size_t ws_size,
                              hipStream_t stream)
{
    const float*     x   = (const float*)    d_in[0];
    const long long* ei  = (const long long*)d_in[1];
    const float*     W1  = (const float*)    d_in[2];
    const float*     b1  = (const float*)    d_in[3];
    const float*     W2  = (const float*)    d_in[4];
    const float*     b2  = (const float*)    d_in[5];
    const float*     Wfc = (const float*)    d_in[6];
    const float*     bfc = (const float*)    d_in[7];
    float*           out = (float*)d_out;

    const long long N = (long long)in_sizes[0] / CH;
    const long long E = (long long)in_sizes[1] / 2;

    char* ws = (char*)d_ws;
    size_t off = 0;
    float* dinv   = (float*)(ws + off); off += ((size_t)N * 4 + 255) & ~(size_t)255;
    float* buf0   = (float*)(ws + off); off += (size_t)N * CH * 4;
    float* buf1   = (float*)(ws + off); off += (size_t)N * CH * 4;
    float* pooled = (float*)(ws + off); off += 256 * 4;
    (void)ws_size; (void)n_in; (void)out_size;

    const int T256 = 256;
    dim3 gN((unsigned)((N + 255) / 256));
    dim3 gE((unsigned)((E + 255) / 256));
    dim3 gEdge((unsigned)((E + 7) / 8));
    dim3 gGemm((unsigned)((N + 127) / 128));
    dim3 gElem((unsigned)((N * 32 + 255) / 256));
    dim3 gPool((unsigned)((N + 255) / 256));
    size_t rowBytes = (size_t)N * CH * 4;

    // normalization: deg = 1 (self loop) + in-degree; dinv = rsqrt(deg)
    k_fill<<<gN, T256, 0, stream>>>(dinv, 1.0f, N);
    k_deg<<<gE, T256, 0, stream>>>(ei, E, dinv);
    k_rsqrt<<<gN, T256, 0, stream>>>(dinv, N);

    // ---- layer 1 ----
    gemm_rowscale_bf16<<<gGemm, T256, 0, stream>>>(x, W1, dinv, buf0, (int)N);
    hipMemcpyAsync(buf1, buf0, rowBytes, hipMemcpyDeviceToDevice, stream); // self loops
    k_scatter<<<gEdge, T256, 0, stream>>>(ei, E, buf0, buf1);
    k_finalize<<<gElem, T256, 0, stream>>>(buf1, dinv, b1, N);             // H1 in buf1

    // ---- layer 2 ----
    gemm_rowscale_bf16<<<gGemm, T256, 0, stream>>>(buf1, W2, dinv, buf0, (int)N);
    hipMemcpyAsync(buf1, buf0, rowBytes, hipMemcpyDeviceToDevice, stream); // self loops
    k_scatter<<<gEdge, T256, 0, stream>>>(ei, E, buf0, buf1);
    k_finalize<<<gElem, T256, 0, stream>>>(buf1, dinv, b2, N);             // H2 in buf1

    // ---- pool + FC ----
    k_fill<<<dim3(1), T256, 0, stream>>>(pooled, 0.0f, 256);
    k_pool<<<gPool, 128, 0, stream>>>(buf1, N, pooled);
    k_fc<<<dim3(1), 128, 0, stream>>>(pooled, Wfc, bfc, out, 1.0f / (float)N);
}
